// EMaDelta_35588099014893
// MI455X (gfx1250) — compile-verified, matrix operand order checked by reading, not run on hardware
//
#include <hip/hip_runtime.h>
#include <stdint.h>

// ---------- problem constants (B=8, L=512, H=256, V=32000) ----------
static constexpr int kB = 8, kL = 512, kH = 256, kH2 = 512, kV = 32000;
static constexpr int kT = kB * kL;            // 4096 tokens
static constexpr float kBeta = 1.0f - 0.95f;  // (1-ALPHA)
static constexpr int kC = 16;                 // scan chunk size
static constexpr int kNC = kL / kC;           // 32 chunks (last one padded)

typedef __attribute__((ext_vector_type(16))) __bf16 v16bf;
typedef __attribute__((ext_vector_type(8)))  float  v8f;
typedef __attribute__((ext_vector_type(4)))  unsigned int u32x4;
typedef __attribute__((ext_vector_type(4)))  float  f32x4;

union Frag { u32x4 u[2]; v16bf v; };

__device__ __forceinline__ uint16_t f2bf(float f) {
  unsigned int u = __float_as_uint(f);
  unsigned int r = u + 0x7FFFu + ((u >> 16) & 1u);   // round-to-nearest-even
  return (uint16_t)(r >> 16);
}
__device__ __forceinline__ float bflo(unsigned int p) { return __uint_as_float(p << 16); }
__device__ __forceinline__ float bfhi(unsigned int p) { return __uint_as_float(p & 0xFFFF0000u); }

// fragment loader: row-major bf16 matrix, 16-bit operand layout
__device__ __forceinline__ v16bf load_frag(const uint16_t* row_ptr, int k0) {
  Frag f;
  f.u[0] = *(const u32x4*)(row_ptr + k0);
  f.u[1] = *(const u32x4*)(row_ptr + k0 + 16);
  return f.v;
}

// ---------- K0: fp32 weights -> transposed bf16 (W1T[N=512][K=256], W2T[N=256][K=512]) ----------
__global__ void k_cvt_weights(const float* __restrict__ W1, const float* __restrict__ W2,
                              uint16_t* __restrict__ W1T, uint16_t* __restrict__ W2T) {
  int i = blockIdx.x * 256 + threadIdx.x;          // grid covers 131072 exactly
  { int n = i >> 8, k = i & 255;  W1T[i] = f2bf(W1[k * kH2 + n]); }
  { int n = i >> 9, k = i & 511;  W2T[i] = f2bf(W2[k * kH  + n]); }
}

// ---------- K1: embedding gather -> fp32 residual copy + bf16 GEMM operand ----------
__global__ void k_gather(const int* __restrict__ seq, const float* __restrict__ embed,
                         float* __restrict__ h32, uint16_t* __restrict__ hb) {
  int t = blockIdx.x, c = threadIdx.x;
  int row = seq[t];
  float v = embed[(size_t)row * kH + c];
  h32[(size_t)t * kH + c] = v;
  hb [(size_t)t * kH + c] = f2bf(v);
}

// ---------- K2: C1[4096,512] = ReLU(Hb @ W1 + b1); 32x64 tile per wave ----------
__global__ void __launch_bounds__(128) k_gemm1(const uint16_t* __restrict__ Hb,
                                               const uint16_t* __restrict__ W1T,
                                               const float* __restrict__ b1,
                                               uint16_t* __restrict__ C1) {
  const int tid = threadIdx.x, wave = tid >> 5, lane = tid & 31;
  const int l15 = lane & 15, lh = lane >> 4;
  const int mbase = blockIdx.x * 32;
  const int nbase = blockIdx.y * 256 + wave * 64;

  v8f acc[2][4] = {};
  const uint16_t* arow0 = Hb + (size_t)(mbase + l15) * kH;
  const uint16_t* arow1 = Hb + (size_t)(mbase + 16 + l15) * kH;
  for (int kk = 0; kk < kH; kk += 32) {
    const int k0 = kk + lh * 8;
    v16bf a0 = load_frag(arow0, k0);
    v16bf a1 = load_frag(arow1, k0);
#pragma unroll
    for (int j = 0; j < 4; ++j) {
      const uint16_t* brow = W1T + (size_t)(nbase + j * 16 + l15) * kH;
      v16bf b = load_frag(brow, k0);
      acc[0][j] = __builtin_amdgcn_wmma_f32_16x16x32_bf16(false, a0, false, b,
                                                          (short)0, acc[0][j], false, false);
      acc[1][j] = __builtin_amdgcn_wmma_f32_16x16x32_bf16(false, a1, false, b,
                                                          (short)0, acc[1][j], false, false);
    }
  }
#pragma unroll
  for (int h = 0; h < 2; ++h)
#pragma unroll
    for (int j = 0; j < 4; ++j) {
      const int col = nbase + j * 16 + l15;
      const float bias = b1[col];
#pragma unroll
      for (int r = 0; r < 8; ++r) {
        const int row = mbase + 16 * h + r + lh * 8;
        float v = acc[h][j][r] + bias;
        v = v > 0.0f ? v : 0.0f;
        C1[(size_t)row * kH2 + col] = f2bf(v);
      }
    }
}

// ---------- K3: X = C1 @ W2 + b2 + h ; fused LayerNorm, key-norms, bf16 kn matrix ----------
__global__ void __launch_bounds__(128) k_gemm2_ln(const uint16_t* __restrict__ C1,
                                                  const uint16_t* __restrict__ W2T,
                                                  const float* __restrict__ b2,
                                                  const float* __restrict__ h32,
                                                  const float* __restrict__ gamma,
                                                  const float* __restrict__ beta,
                                                  float* __restrict__ hn,
                                                  float* __restrict__ rnorm,
                                                  uint16_t* __restrict__ knb) {
  __shared__ float xt[32][kH + 1];
  const int tid = threadIdx.x, wave = tid >> 5, lane = tid & 31;
  const int l15 = lane & 15, lh = lane >> 4;
  const int mbase = blockIdx.x * 32;
  const int nbase = wave * 64;

  v8f acc[2][4] = {};
  const uint16_t* arow0 = C1 + (size_t)(mbase + l15) * kH2;
  const uint16_t* arow1 = C1 + (size_t)(mbase + 16 + l15) * kH2;
  for (int kk = 0; kk < kH2; kk += 32) {
    const int k0 = kk + lh * 8;
    v16bf a0 = load_frag(arow0, k0);
    v16bf a1 = load_frag(arow1, k0);
#pragma unroll
    for (int j = 0; j < 4; ++j) {
      const uint16_t* brow = W2T + (size_t)(nbase + j * 16 + l15) * kH2;
      v16bf b = load_frag(brow, k0);
      acc[0][j] = __builtin_amdgcn_wmma_f32_16x16x32_bf16(false, a0, false, b,
                                                          (short)0, acc[0][j], false, false);
      acc[1][j] = __builtin_amdgcn_wmma_f32_16x16x32_bf16(false, a1, false, b,
                                                          (short)0, acc[1][j], false, false);
    }
  }
#pragma unroll
  for (int h = 0; h < 2; ++h)
#pragma unroll
    for (int j = 0; j < 4; ++j) {
      const int col = nbase + j * 16 + l15;
      const float bias = b2[col];
#pragma unroll
      for (int r = 0; r < 8; ++r) {
        const int rl = 16 * h + r + lh * 8;
        xt[rl][col] = acc[h][j][r] + bias + h32[(size_t)(mbase + rl) * kH + col];
      }
    }
  __syncthreads();

  // LayerNorm: one wave per 8 rows; also emit row norm + bf16 normalized key row
  for (int rr = 0; rr < 8; ++rr) {
    const int row = wave * 8 + rr;
    const int token = mbase + row;
    float xv[8], s1 = 0.0f, s2 = 0.0f;
#pragma unroll
    for (int i = 0; i < 8; ++i) {
      xv[i] = xt[row][lane + 32 * i];
      s1 += xv[i];
      s2 += xv[i] * xv[i];
    }
    for (int off = 16; off; off >>= 1) {
      s1 += __shfl_xor(s1, off, 32);
      s2 += __shfl_xor(s2, off, 32);
    }
    const float mu = s1 * (1.0f / kH);
    const float var = s2 * (1.0f / kH) - mu * mu;
    const float rstd = rsqrtf(var + 1e-5f);
    float yv[8], s3 = 0.0f;
#pragma unroll
    for (int i = 0; i < 8; ++i) {
      const int c = lane + 32 * i;
      yv[i] = (xv[i] - mu) * rstd * gamma[c] + beta[c];
      s3 += yv[i] * yv[i];
      hn[(size_t)token * kH + c] = yv[i];
    }
    for (int off = 16; off; off >>= 1) s3 += __shfl_xor(s3, off, 32);
    const float rn = sqrtf(s3);                       // all lanes hold full sum
    const float ri = 1.0f / fmaxf(rn, 1e-12f);
    const bool last = ((token & (kL - 1)) == (kL - 1));  // query token: excluded from scan
#pragma unroll
    for (int i = 0; i < 8; ++i) {
      const int c = lane + 32 * i;
      knb[(size_t)token * kH + c] = last ? (uint16_t)0 : f2bf(yv[i] * ri);
    }
    if (lane == 0) rnorm[token] = rn;
  }
}

// ---------- K4: chunked delta-rule backward scan (WMMA + 16-step triangular solve) ----------
// Per chunk of C=16 steps:  G = Kn Kn^T (WMMA),  c = Kn u (WMMA, u broadcast to 16 cols),
// solve s_t = c_t - beta * sum_{s>t} G[t,s] s_s  (t = 15..0, 2 shfls/step),
// then u -= beta * Kn^T s ,  m += beta * K^T s  (K = Kn * rownorm).
__global__ void __launch_bounds__(32) k_scan(const float* __restrict__ hn,
                                             const uint16_t* __restrict__ knb,
                                             const float* __restrict__ rnorm,
                                             float* __restrict__ mout) {
  __shared__ alignas(16) uint16_t ubf[kH];     // bf16 copy of u, broadcast B operand
  const int b = blockIdx.x, lane = threadIdx.x;
  const int l15 = lane & 15, lh = lane >> 4;
  const int bbase = b * kL;

  float u[8], m[8];
  {  // u = q = last token's normalized hidden state (lane owns dims 8*lane..8*lane+7)
    const float* q = hn + ((size_t)bbase + kL - 1) * kH + lane * 8;
    f32x4 q0 = *(const f32x4*)(q);
    f32x4 q1 = *(const f32x4*)(q + 4);
#pragma unroll
    for (int i = 0; i < 4; ++i) { u[i] = q0[i]; u[4 + i] = q1[i]; m[i] = 0.0f; m[4 + i] = 0.0f; }
  }

  for (int c = kNC - 1; c >= 0; --c) {
    const int t0 = c * kC;
    // ---- stage bf16 u into LDS (one b128 store per lane) ----
    __syncthreads();
    {
      unsigned int p[4];
#pragma unroll
      for (int w = 0; w < 4; ++w)
        p[w] = (unsigned int)f2bf(u[2 * w]) | ((unsigned int)f2bf(u[2 * w + 1]) << 16);
      *(u32x4*)(ubf + lane * 8) = u32x4{p[0], p[1], p[2], p[3]};
    }
    __syncthreads();

    // ---- G = Kn Kn^T and cvec = Kn u via WMMA (A frag == B frag for G) ----
    v8f Gacc = {}, Cacc = {};
    const uint16_t* arow = knb + (size_t)(bbase + t0 + l15) * kH;
    for (int kk = 0; kk < kH; kk += 32) {
      const int k0 = kk + lh * 8;
      v16bf a = load_frag(arow, k0);
      Frag bu;
      bu.u[0] = *(const u32x4*)(ubf + k0);
      bu.u[1] = *(const u32x4*)(ubf + k0 + 16);
      Gacc = __builtin_amdgcn_wmma_f32_16x16x32_bf16(false, a, false, a,
                                                     (short)0, Gacc, false, false);
      Cacc = __builtin_amdgcn_wmma_f32_16x16x32_bf16(false, a, false, bu.v,
                                                     (short)0, Cacc, false, false);
    }

    // ---- preload this chunk's 16 kn rows (packed bf16) + row norms ----
    u32x4 kr[kC];
    float rn[kC];
#pragma unroll
    for (int t = 0; t < kC; ++t) {
      kr[t] = *(const u32x4*)(knb + (size_t)(bbase + t0 + t) * kH + lane * 8);
      rn[t] = rnorm[bbase + t0 + t];
    }
    if (c > 0)  // warm L0 for next chunk's fragments
      __builtin_prefetch(knb + (size_t)(bbase + t0 - 16 + l15) * kH + lh * 64, 0, 0);

    // ---- accS = c_{l15}: extract from Cacc (cols replicated; rows split by lane half) ----
    float cand = Cacc[0];
#pragma unroll
    for (int r = 1; r < 8; ++r) cand = ((l15 & 7) == r) ? Cacc[r] : cand;
    float other = __shfl_xor(cand, 16, 32);
    float accS = ((l15 >> 3) == lh) ? cand : other;

    // ---- triangular solve, t = 15..0 (2 shfls per step) ----
    float s[kC];
#pragma unroll
    for (int t = kC - 1; t >= 0; --t) {
      const float sb = __shfl(accS, t, 32);       // s_t from lane t
      s[t] = sb;
      const float gc = Gacc[t & 7];               // G[t][l15] lives in half (t>>3)
      const float go = __shfl_xor(gc, 16, 32);
      const float gv = ((t >> 3) == lh) ? gc : go;
      accS -= kBeta * gv * sb;                    // only lanes l15 < t still matter
    }

    // ---- rank-16 update: u -= beta*Kn^T s ; m += beta*K^T s ----
#pragma unroll
    for (int t = 0; t < kC; ++t) {
      const float su = kBeta * s[t];
      const float sm = su * rn[t];
      const u32x4 kv = kr[t];
#pragma unroll
      for (int w = 0; w < 4; ++w) {
        const float k0f = bflo(kv[w]);
        const float k1f = bfhi(kv[w]);
        u[2 * w]     -= su * k0f;  m[2 * w]     += sm * k0f;
        u[2 * w + 1] -= su * k1f;  m[2 * w + 1] += sm * k1f;
      }
    }
  }

  float* mo = mout + (size_t)b * kH + lane * 8;
  *(f32x4*)(mo)     = f32x4{m[0], m[1], m[2], m[3]};
  *(f32x4*)(mo + 4) = f32x4{m[4], m[5], m[6], m[7]};
}

// ---------- K5: tmp = m @ Wrp + brp ----------
__global__ void __launch_bounds__(256) k_readproj(const float* __restrict__ m,
                                                  const float* __restrict__ Wrp,
                                                  const float* __restrict__ brp,
                                                  float* __restrict__ tmp) {
  __shared__ float ms[kB * kH];
  const int tid = threadIdx.x;
  for (int i = tid; i < kB * kH; i += 256) ms[i] = m[i];
  __syncthreads();
  float acc[kB] = {};
  for (int h = 0; h < kH; ++h) {
    const float w = Wrp[(size_t)h * kH + tid];
#pragma unroll
    for (int b = 0; b < kB; ++b) acc[b] += ms[b * kH + h] * w;
  }
  const float bb = brp[tid];
#pragma unroll
  for (int b = 0; b < kB; ++b) tmp[b * kH + tid] = acc[b] + bb;
}

// ---------- K6: out = tmp @ Wout + bout  (bandwidth bound: 32.8 MB of Wout, read once) ----------
__global__ void __launch_bounds__(256) k_out(const float* __restrict__ tmp,
                                             const float* __restrict__ Wout,
                                             const float* __restrict__ bout,
                                             float* __restrict__ out) {
  __shared__ float ts[kB * kH];
  const int tid = threadIdx.x;
  for (int i = tid; i < kB * kH; i += 256) ts[i] = tmp[i];
  __syncthreads();
  const int v = blockIdx.x * 256 + tid;
  float acc[kB] = {};
  for (int h = 0; h < kH; ++h) {
    const float w = Wout[(size_t)h * kV + v];
#pragma unroll
    for (int b = 0; b < kB; ++b) acc[b] += ts[b * kH + h] * w;
  }
  const float bb = bout[v];
#pragma unroll
  for (int b = 0; b < kB; ++b) out[(size_t)b * kV + v] = acc[b] + bb;
}

extern "C" void kernel_launch(void* const* d_in, const int* in_sizes, int n_in,
                              void* d_out, int out_size, void* d_ws, size_t ws_size,
                              hipStream_t stream) {
  const int*   seq   = (const int*)  d_in[0];
  const float* embed = (const float*)d_in[1];
  const float* W1    = (const float*)d_in[2];
  const float* b1    = (const float*)d_in[3];
  const float* W2    = (const float*)d_in[4];
  const float* b2    = (const float*)d_in[5];
  const float* gamma = (const float*)d_in[6];
  const float* beta  = (const float*)d_in[7];
  const float* Wrp   = (const float*)d_in[8];
  const float* brp   = (const float*)d_in[9];
  const float* Wout  = (const float*)d_in[10];
  const float* bout  = (const float*)d_in[11];
  float* out = (float*)d_out;

  char* ws = (char*)d_ws;
  size_t off = 0;
  auto alloc = [&](size_t bytes) -> void* {
    void* p = ws + off;
    off += (bytes + 255) & ~(size_t)255;
    return p;
  };
  uint16_t* W1T = (uint16_t*)alloc((size_t)kH * kH2 * 2);
  uint16_t* W2T = (uint16_t*)alloc((size_t)kH * kH2 * 2);
  uint16_t* Hb  = (uint16_t*)alloc((size_t)kT * kH * 2);
  float*    h32 = (float*)   alloc((size_t)kT * kH * 4);
  uint16_t* C1  = (uint16_t*)alloc((size_t)kT * kH2 * 2);
  float*    hn  = (float*)   alloc((size_t)kT * kH * 4);
  uint16_t* knb = (uint16_t*)alloc((size_t)kT * kH * 2);
  float*    rno = (float*)   alloc((size_t)kT * 4);
  float*    mv  = (float*)   alloc((size_t)kB * kH * 4);
  float*    tv  = (float*)   alloc((size_t)kB * kH * 4);

  k_cvt_weights<<<(kH * kH2) / 256, 256, 0, stream>>>(W1, W2, W1T, W2T);
  k_gather<<<kT, kH, 0, stream>>>(seq, embed, h32, Hb);
  dim3 g1(kT / 32, kH2 / 256);
  k_gemm1<<<g1, 128, 0, stream>>>(Hb, W1T, b1, C1);
  k_gemm2_ln<<<kT / 32, 128, 0, stream>>>(C1, W2T, b2, h32, gamma, beta, hn, rno, knb);
  k_scan<<<kB, 32, 0, stream>>>(hn, knb, rno, mv);
  k_readproj<<<1, 256, 0, stream>>>(mv, Wrp, brp, tv);
  k_out<<<kV / 256, 256, 0, stream>>>(tv, Wout, bout, out);
}
